// XLNetModel_60086592471113
// MI455X (gfx1250) — compile-verified, matrix operand order checked by reading, not run on hardware
//
#include <hip/hip_runtime.h>

#define D_   1024
#define NH_  16
#define DH_  64
#define DI_  4096
#define NL_  4
#define B_   4
#define L_   512
#define M_   64
#define SCALE_ 0.125f
#define NEG_   1e30f
#define EPS_   1e-8f

typedef __bf16 v16bf __attribute__((ext_vector_type(16)));
typedef float  v8f   __attribute__((ext_vector_type(8)));

__device__ __forceinline__ __bf16 f2bf(float x) { return (__bf16)x; }

__device__ __forceinline__ v8f wmma_bf16(v16bf a, v16bf b, v8f c) {
  // D = A(16x32 bf16) * B(32x16 bf16) + C(16x16 f32)
  return __builtin_amdgcn_wmma_f32_16x16x32_bf16(false, a, false, b, (short)0, c,
                                                 false, false);
}

// Async DMA: 16 bytes global -> LDS, tracked by ASYNCcnt (CDNA5 §10.6 async ops).
// LDS aperture maps addr[31:0] directly, so the low 32 bits of a generic pointer
// into __shared__ are the LDS byte offset.
__device__ __forceinline__ void async_copy_b128(void* lds_dst, const void* gsrc) {
  unsigned int loff = (unsigned int)(unsigned long long)lds_dst;
  asm volatile("global_load_async_to_lds_b128 %0, %1, off"
               :
               : "v"(loff), "v"(gsrc)
               : "memory");
}

__device__ __forceinline__ void wait_async0() {
#if __has_builtin(__builtin_amdgcn_s_wait_asynccnt)
  __builtin_amdgcn_s_wait_asynccnt(0);
#else
  asm volatile("s_wait_asynccnt 0" ::: "memory");
#endif
}

// A fragment: element (m,k) of a 16x32 row-major tile, row stride ldm (floats).
// Lane layout (ISA 7.12.2): lanes 0-15 m=lane K={0..7,16..23}; lanes16-31 K={8..15,24..31}
__device__ __forceinline__ v16bf ldfragA(const float* p, int ldm) {
  int lane = threadIdx.x & 31;
  int m = lane & 15;
  int kg = (lane >> 4) * 8;
  const float* row = p + (long)m * ldm;
  v16bf a;
#pragma unroll
  for (int v = 0; v < 4; ++v) {
    a[2 * v]     = f2bf(row[kg + 2 * v]);
    a[2 * v + 1] = f2bf(row[kg + 2 * v + 1]);
    a[8 + 2 * v]     = f2bf(row[16 + kg + 2 * v]);
    a[8 + 2 * v + 1] = f2bf(row[16 + kg + 2 * v + 1]);
  }
  return a;
}

// Same as ldfragA but adds a per-k bias vector (q + r_w_bias / q + r_r_bias)
__device__ __forceinline__ v16bf ldfragA_bias(const float* p, int ldm, const float* bias) {
  int lane = threadIdx.x & 31;
  int m = lane & 15;
  int kg = (lane >> 4) * 8;
  const float* row = p + (long)m * ldm;
  v16bf a;
#pragma unroll
  for (int v = 0; v < 4; ++v) {
    a[2 * v]     = f2bf(row[kg + 2 * v] + bias[kg + 2 * v]);
    a[2 * v + 1] = f2bf(row[kg + 2 * v + 1] + bias[kg + 2 * v + 1]);
    a[8 + 2 * v]     = f2bf(row[16 + kg + 2 * v] + bias[16 + kg + 2 * v]);
    a[8 + 2 * v + 1] = f2bf(row[16 + kg + 2 * v + 1] + bias[16 + kg + 2 * v + 1]);
  }
  return a;
}

// B fragment from a row-major [K=32, N=16] tile, row stride ldk (standard C = A*B)
__device__ __forceinline__ v16bf ldfragB(const float* p, int ldk) {
  int lane = threadIdx.x & 31;
  int n = lane & 15;
  int kg = (lane >> 4) * 8;
  v16bf b;
#pragma unroll
  for (int v = 0; v < 4; ++v) {
    b[2 * v]     = f2bf(p[(long)(kg + 2 * v) * ldk + n]);
    b[2 * v + 1] = f2bf(p[(long)(kg + 2 * v + 1) * ldk + n]);
    b[8 + 2 * v]     = f2bf(p[(long)(16 + kg + 2 * v) * ldk + n]);
    b[8 + 2 * v + 1] = f2bf(p[(long)(16 + kg + 2 * v + 1) * ldk + n]);
  }
  return b;
}
// NOTE: loading a row-major 16xK tile X with ldfragA *as the B operand* presents X^T,
// i.e. computes A * X^T  (transpose symmetry of the A/B lane layouts).

// ---------------------------------------------------------------------------
// Generic GEMM: C[M,N] = act( A[M,K] * B[K,N] + bias ).  fp32 in/out, bf16 WMMA,
// fp32 accum. Block = 256 thr (8 waves), tile 64x64x32, double-buffered
// async-to-LDS copies overlapped with the WMMA loop.
// ---------------------------------------------------------------------------
__global__ __launch_bounds__(256) void gemm_kernel(
    const float* __restrict__ A, const float* __restrict__ Bm,
    float* __restrict__ C, const float* __restrict__ bias,
    int K, int N, int act) {
  __shared__ __align__(16) float As[2][64 * 36];  // row stride 144B (16B aligned)
  __shared__ __align__(16) float Bs[2][32 * 68];  // row stride 272B (16B aligned)
  int tid  = threadIdx.x;
  int wave = tid >> 5;
  int lane = tid & 31;
  int bm = blockIdx.y * 64;
  int bn = blockIdx.x * 64;
  int mt  = wave >> 1;        // M sub-tile 0..3
  int nt0 = (wave & 1) * 2;   // N sub-tiles nt0, nt0+1

  // per-thread DMA assignments: 2 x b128 for A (64x32), 2 x b128 for B (32x64)
  int ra0 = tid >> 3,          ca0 = (tid & 7) * 4;
  int ra1 = (tid + 256) >> 3,  ca1 = ((tid + 256) & 7) * 4;
  int rb0 = tid >> 4,          cb0 = (tid & 15) * 4;
  int rb1 = (tid + 256) >> 4,  cb1 = ((tid + 256) & 15) * 4;

  auto issue_tile = [&](int buf, int k0) {
    async_copy_b128(&As[buf][ra0 * 36 + ca0], &A[(long)(bm + ra0) * K + k0 + ca0]);
    async_copy_b128(&As[buf][ra1 * 36 + ca1], &A[(long)(bm + ra1) * K + k0 + ca1]);
    async_copy_b128(&Bs[buf][rb0 * 68 + cb0], &Bm[(long)(k0 + rb0) * N + bn + cb0]);
    async_copy_b128(&Bs[buf][rb1 * 68 + cb1], &Bm[(long)(k0 + rb1) * N + bn + cb1]);
  };

  v8f acc0 = {0.f, 0.f, 0.f, 0.f, 0.f, 0.f, 0.f, 0.f};
  v8f acc1 = {0.f, 0.f, 0.f, 0.f, 0.f, 0.f, 0.f, 0.f};

  issue_tile(0, 0);
  wait_async0();
  __syncthreads();

  int nsteps = K >> 5;
  for (int s = 0; s < nsteps; ++s) {
    int cur = s & 1;
    if (s + 1 < nsteps) issue_tile(cur ^ 1, (s + 1) * 32);  // overlap DMA w/ WMMA
    v16bf af  = ldfragA(&As[cur][(mt * 16) * 36], 36);
    v16bf bf0 = ldfragB(&Bs[cur][nt0 * 16], 68);
    v16bf bf1 = ldfragB(&Bs[cur][(nt0 + 1) * 16], 68);
    acc0 = wmma_bf16(af, bf0, acc0);
    acc1 = wmma_bf16(af, bf1, acc1);
    wait_async0();
    __syncthreads();
  }

  int n  = lane & 15;
  int mb = (lane >> 4) * 8;
#pragma unroll
  for (int r = 0; r < 8; ++r) {
    int gm = bm + mt * 16 + mb + r;
#pragma unroll
    for (int t = 0; t < 2; ++t) {
      int gn = bn + (nt0 + t) * 16 + n;
      float vv = (t == 0) ? acc0[r] : acc1[r];
      if (bias) vv += bias[gn];
      if (act == 1) vv = 0.5f * vv * (1.f + erff(vv * 0.70710678118f));  // exact GELU
      C[(long)gm * N + gn] = vv;
    }
  }
}

// ---------------------------------------------------------------------------
// Relative attention core for one (b, head, 16-query tile).
// scores kept in LDS (16 x 512 f32); rel_shift applied on scatter:
//   bd_shifted[i,j] = bd_raw[i, j + 512 - i]
// ---------------------------------------------------------------------------
__global__ __launch_bounds__(256) void attn_kernel(
    const float* __restrict__ qsrc,  // [L*B, NH*DH]
    const float* __restrict__ kh,    // [L*B, NH*DH]
    const float* __restrict__ vh,    // [L*B, NH*DH]
    const float* __restrict__ kr,    // [2L, NH*DH] (batch-broadcast)
    const float* __restrict__ rw,    // [NH*DH]  r_w_bias (layer)
    const float* __restrict__ rr,    // [NH*DH]  r_r_bias (layer)
    const float* __restrict__ perm,  // [B, L, L]
    float* __restrict__ out,         // [L*B, NH*DH]
    int use_eye) {
  __shared__ float s[16 * 512];
  int i0 = blockIdx.x * 16;
  int nh = blockIdx.y;
  int b  = blockIdx.z;
  int tid = threadIdx.x, wave = tid >> 5, lane = tid & 31;
  const int LDQ = B_ * (NH_ * DH_);  // row stride over time index
  const float* qp = qsrc + ((long)i0 * B_ + b) * (NH_ * DH_) + nh * DH_;
  const float* kp = kh + (long)b * (NH_ * DH_) + nh * DH_;
  const float* vp = vh + (long)b * (NH_ * DH_) + nh * DH_;
  const float* rp = kr + nh * DH_;
  const float* rwp = rw + nh * DH_;
  const float* rrp = rr + nh * DH_;
  int n  = lane & 15;
  int mb = (lane >> 4) * 8;

  // ---- content term: ac[i,j] = (q+rw) . kh[j]  (j = 0..511) -------------
  v16bf aw0 = ldfragA_bias(qp, LDQ, rwp);
  v16bf aw1 = ldfragA_bias(qp + 32, LDQ, rwp + 32);
  for (int jt = wave; jt < 32; jt += 8) {
    int j0 = jt * 16;
    v8f acc = {0.f, 0.f, 0.f, 0.f, 0.f, 0.f, 0.f, 0.f};
    v16bf bk0 = ldfragA(kp + (long)j0 * LDQ, LDQ);        // kh tile as B => q*kh^T
    v16bf bk1 = ldfragA(kp + (long)j0 * LDQ + 32, LDQ);
    acc = wmma_bf16(aw0, bk0, acc);
    acc = wmma_bf16(aw1, bk1, acc);
#pragma unroll
    for (int r = 0; r < 8; ++r) s[(mb + r) * 512 + j0 + n] = acc[r];
  }
  __syncthreads();

  // ---- positional term with rel_shift: bd[i,jr] = (q+rr) . kr[jr] --------
  v16bf ar0 = ldfragA_bias(qp, LDQ, rrp);
  v16bf ar1 = ldfragA_bias(qp + 32, LDQ, rrp + 32);
  for (int jt = wave; jt < 64; jt += 8) {
    int j0 = jt * 16;
    v8f acc = {0.f, 0.f, 0.f, 0.f, 0.f, 0.f, 0.f, 0.f};
    v16bf br0 = ldfragA(rp + (long)j0 * (NH_ * DH_), NH_ * DH_);
    v16bf br1 = ldfragA(rp + (long)j0 * (NH_ * DH_) + 32, NH_ * DH_);
    acc = wmma_bf16(ar0, br0, acc);
    acc = wmma_bf16(ar1, br1, acc);
#pragma unroll
    for (int r = 0; r < 8; ++r) {
      int m = mb + r;
      int i = i0 + m;
      int jcol = j0 + n - 512 + i;  // inverse of rel_shift
      if (jcol >= 0 && jcol < 512) s[m * 512 + jcol] += acc[r];
    }
  }
  __syncthreads();

  // ---- mask + softmax over j (each wave handles 2 rows) ------------------
#pragma unroll
  for (int rr2 = 0; rr2 < 2; ++rr2) {
    int m = wave * 2 + rr2;
    int i = i0 + m;
    const float* pm = perm + ((long)b * L_ + i) * L_;
    float vals[16];
    float mx = -3.4e38f;
#pragma unroll
    for (int t = 0; t < 16; ++t) {
      int c = lane + t * 32;
      float msk = (pm[c] > 0.f && !(use_eye && i == c)) ? 1.f : 0.f;
      float sc = s[m * 512 + c] * SCALE_ - NEG_ * msk;
      vals[t] = sc;
      mx = fmaxf(mx, sc);
    }
    for (int off = 16; off > 0; off >>= 1) mx = fmaxf(mx, __shfl_xor(mx, off, 32));
    float sum = 0.f;
#pragma unroll
    for (int t = 0; t < 16; ++t) { vals[t] = __expf(vals[t] - mx); sum += vals[t]; }
    for (int off = 16; off > 0; off >>= 1) sum += __shfl_xor(sum, off, 32);
    float inv = 1.f / sum;
#pragma unroll
    for (int t = 0; t < 16; ++t) s[m * 512 + lane + t * 32] = vals[t] * inv;
  }
  __syncthreads();

  // ---- out[i,d] = prob[i,:] * vh[:,d]  (waves 0..3, one 16-wide d tile) --
  if (wave < 4) {
    v8f acc = {0.f, 0.f, 0.f, 0.f, 0.f, 0.f, 0.f, 0.f};
    for (int k0 = 0; k0 < 512; k0 += 32) {
      v16bf af = ldfragA(&s[k0], 512);                       // prob tile
      v16bf bf = ldfragB(vp + (long)k0 * LDQ + wave * 16, LDQ);
      acc = wmma_bf16(af, bf, acc);
    }
    float* op = out + ((long)i0 * B_ + b) * (NH_ * DH_) + nh * DH_ + wave * 16;
#pragma unroll
    for (int r = 0; r < 8; ++r) op[(long)(mb + r) * LDQ + n] = acc[r];
  }
}

// ---------------------------------------------------------------------------
// LayerNorm(x + res) with per-row block and cross-wave LDS reduction
// ---------------------------------------------------------------------------
__global__ __launch_bounds__(256) void ln_kernel(
    const float* __restrict__ x, const float* __restrict__ res,
    const float* __restrict__ w, const float* __restrict__ bb,
    float* __restrict__ o) {
  __shared__ float red[8];
  int row = blockIdx.x, tid = threadIdx.x;
  const float* xp = x + (long)row * D_;
  const float* rp = res + (long)row * D_;
  float v[4];
  float sum = 0.f;
#pragma unroll
  for (int t = 0; t < 4; ++t) { v[t] = xp[tid + t * 256] + rp[tid + t * 256]; sum += v[t]; }
  for (int off = 16; off > 0; off >>= 1) sum += __shfl_xor(sum, off, 32);
  if ((tid & 31) == 0) red[tid >> 5] = sum;
  __syncthreads();
  float tot = 0.f;
#pragma unroll
  for (int i = 0; i < 8; ++i) tot += red[i];
  float mean = tot / (float)D_;
  __syncthreads();
  float vs = 0.f;
#pragma unroll
  for (int t = 0; t < 4; ++t) { float d = v[t] - mean; vs += d * d; }
  for (int off = 16; off > 0; off >>= 1) vs += __shfl_xor(vs, off, 32);
  if ((tid & 31) == 0) red[tid >> 5] = vs;
  __syncthreads();
  float tv = 0.f;
#pragma unroll
  for (int i = 0; i < 8; ++i) tv += red[i];
  float rstd = rsqrtf(tv / (float)D_ + EPS_);
#pragma unroll
  for (int t = 0; t < 4; ++t) {
    int c = tid + t * 256;
    o[(long)row * D_ + c] = (v[t] - mean) * rstd * w[c] + bb[c];
  }
}

// qg[m,b,:] -> qgl[l,b,:] via one-hot target mapping (scatter as weighted sum)
__global__ __launch_bounds__(256) void scatter_tm(
    const float* __restrict__ qg, const float* __restrict__ tmap,
    float* __restrict__ qgl) {
  int row = blockIdx.x;           // l*B + b
  int l = row >> 2, b = row & 3;
  int tid = threadIdx.x;
  const float* tp = tmap + (long)b * M_ * L_ + l;
  float a0 = 0, a1 = 0, a2 = 0, a3 = 0;
  for (int m = 0; m < M_; ++m) {
    float w = tp[(long)m * L_];
    if (w != 0.f) {
      const float* qp = qg + ((long)m * B_ + b) * D_;
      a0 += w * qp[tid];       a1 += w * qp[tid + 256];
      a2 += w * qp[tid + 512]; a3 += w * qp[tid + 768];
    }
  }
  float* op = qgl + (long)row * D_;
  op[tid] = a0; op[tid + 256] = a1; op[tid + 512] = a2; op[tid + 768] = a3;
}

// avgl[l,b,:] -> avg[m,b,:] via one-hot target mapping (gather)
__global__ __launch_bounds__(256) void gather_tm(
    const float* __restrict__ avgl, const float* __restrict__ tmap,
    float* __restrict__ avg) {
  int row = blockIdx.x;           // m*B + b
  int m = row >> 2, b = row & 3;
  int tid = threadIdx.x;
  const float* tp = tmap + ((long)b * M_ + m) * L_;
  float a0 = 0, a1 = 0, a2 = 0, a3 = 0;
  for (int l = 0; l < L_; ++l) {
    float w = tp[l];
    if (w != 0.f) {
      const float* ap = avgl + ((long)l * B_ + b) * D_;
      a0 += w * ap[tid];       a1 += w * ap[tid + 256];
      a2 += w * ap[tid + 512]; a3 += w * ap[tid + 768];
    }
  }
  float* op = avg + (long)row * D_;
  op[tid] = a0; op[tid + 256] = a1; op[tid + 512] = a2; op[tid + 768] = a3;
}

__global__ void emb_kernel(const int* __restrict__ ids, const float* __restrict__ emb,
                           float* __restrict__ h) {
  int row = blockIdx.x;  // i*B + b (time-major)
  int i = row >> 2, b = row & 3;
  int id = ids[b * L_ + i];
  const float* src = emb + (long)id * D_;
  float* dst = h + (long)row * D_;
  for (int t = threadIdx.x; t < D_; t += 256) dst[t] = src[t];
}

__global__ void ginit_kernel(const float* __restrict__ me, float* __restrict__ g) {
  float* dst = g + (long)blockIdx.x * D_;
  for (int t = threadIdx.x; t < D_; t += 256) dst[t] = me[t];
}

__global__ void posemb_kernel(float* __restrict__ r) {
  int p = blockIdx.x;                     // 0..1023
  float pos = (float)(L_ - p);            // pos_seq = 512, 511, ..., -511
  for (int t = threadIdx.x; t < 512; t += 256) {
    float invf = __powf(10000.f, -(float)(2 * t) / (float)D_);
    float a = pos * invf;
    r[(long)p * D_ + t] = __sinf(a);
    r[(long)p * D_ + 512 + t] = __cosf(a);
  }
}

__global__ void outT_kernel(const float* __restrict__ g, float* __restrict__ o) {
  int row = blockIdx.x;  // m*B + b
  int m = row >> 2, b = row & 3;
  const float* src = g + (long)row * D_;
  float* dst = o + ((long)b * M_ + m) * D_;
  for (int t = threadIdx.x; t < D_; t += 256) dst[t] = src[t];
}

extern "C" void kernel_launch(void* const* d_in, const int* in_sizes, int n_in,
                              void* d_out, int out_size, void* d_ws, size_t ws_size,
                              hipStream_t stream) {
  (void)in_sizes; (void)n_in; (void)out_size; (void)ws_size;
  const int*   ids  = (const int*)d_in[0];
  const float* perm = (const float*)d_in[1];
  const float* tmap = (const float*)d_in[2];
  const float* emb  = (const float*)d_in[3];
  const float* me   = (const float*)d_in[4];
  const float* Wq   = (const float*)d_in[5];
  const float* Wk   = (const float*)d_in[6];
  const float* Wv   = (const float*)d_in[7];
  const float* Wr   = (const float*)d_in[8];
  const float* rrb  = (const float*)d_in[9];   // r_r_bias
  const float* rwb  = (const float*)d_in[10];  // r_w_bias
  const float* ln1w = (const float*)d_in[11];
  const float* ln1b = (const float*)d_in[12];
  const float* fw1  = (const float*)d_in[13];
  const float* fb1  = (const float*)d_in[14];
  const float* fw2  = (const float*)d_in[15];
  const float* fb2  = (const float*)d_in[16];
  const float* ln2w = (const float*)d_in[17];
  const float* ln2b = (const float*)d_in[18];
  float* out = (float*)d_out;

  float* ws = (float*)d_ws;
  size_t off = 0;
  auto alloc = [&](size_t nfl) { float* p = ws + off; off += nfl; return p; };
  const size_t TH = (size_t)L_ * B_ * D_;   // 2048 x 1024
  const size_t TG = (size_t)M_ * B_ * D_;   // 256 x 1024
  float* h    = alloc(TH);
  float* g    = alloc(TG);
  float* r    = alloc((size_t)2 * L_ * D_);
  float* kh   = alloc(TH);
  float* vh   = alloc(TH);
  float* qh   = alloc(TH);
  float* kr   = alloc((size_t)2 * L_ * D_);
  float* qg   = alloc(TG);
  float* qgl  = alloc(TH);
  float* avh  = alloc(TH);
  float* avgl = alloc(TH);
  float* avg  = alloc(TG);
  float* oh   = alloc(TH);
  float* og   = alloc(TG);
  float* f1h  = alloc((size_t)L_ * B_ * DI_);
  float* f1g  = alloc((size_t)M_ * B_ * DI_);
  float* f2h  = alloc(TH);
  float* f2g  = alloc(TG);

  emb_kernel<<<L_ * B_, 256, 0, stream>>>(ids, emb, h);
  ginit_kernel<<<M_ * B_, 256, 0, stream>>>(me, g);
  posemb_kernel<<<2 * L_, 256, 0, stream>>>(r);

  for (int l = 0; l < NL_; ++l) {
    const float* wq = Wq + (size_t)l * D_ * D_;
    const float* wk = Wk + (size_t)l * D_ * D_;
    const float* wv = Wv + (size_t)l * D_ * D_;
    const float* wr = Wr + (size_t)l * D_ * D_;
    const float* rw = rwb + l * NH_ * DH_;
    const float* rrp = rrb + l * NH_ * DH_;
    const float* l1w = ln1w + l * D_;
    const float* l1b = ln1b + l * D_;
    const float* w1 = fw1 + (size_t)l * D_ * DI_;
    const float* b1 = fb1 + (size_t)l * DI_;
    const float* w2 = fw2 + (size_t)l * DI_ * D_;
    const float* b2 = fb2 + (size_t)l * D_;
    const float* l2w = ln2w + l * D_;
    const float* l2b = ln2b + l * D_;

    dim3 gproj(D_ / 64, (L_ * B_) / 64);
    gemm_kernel<<<gproj, 256, 0, stream>>>(h, wk, kh, nullptr, D_, D_, 0);
    gemm_kernel<<<gproj, 256, 0, stream>>>(h, wv, vh, nullptr, D_, D_, 0);
    gemm_kernel<<<gproj, 256, 0, stream>>>(h, wq, qh, nullptr, D_, D_, 0);
    gemm_kernel<<<dim3(D_ / 64, (2 * L_) / 64), 256, 0, stream>>>(r, wr, kr, nullptr, D_, D_, 0);

    // content stream attention (non-target mask includes eye exclusion)
    attn_kernel<<<dim3(L_ / 16, NH_, B_), 256, 0, stream>>>(qh, kh, vh, kr, rw, rrp, perm, avh, 1);
    ln_kernel<<<L_ * B_, 256, 0, stream>>>(avh, h, l1w, l1b, oh);

    // query stream attention (scatter q via target mapping, gather output back)
    gemm_kernel<<<dim3(D_ / 64, (M_ * B_) / 64), 256, 0, stream>>>(g, wq, qg, nullptr, D_, D_, 0);
    scatter_tm<<<L_ * B_, 256, 0, stream>>>(qg, tmap, qgl);
    attn_kernel<<<dim3(L_ / 16, NH_, B_), 256, 0, stream>>>(qgl, kh, vh, kr, rw, rrp, perm, avgl, 0);
    gather_tm<<<M_ * B_, 256, 0, stream>>>(avgl, tmap, avg);
    ln_kernel<<<M_ * B_, 256, 0, stream>>>(avg, g, l1w, l1b, og);

    // FFN (GELU fused into first GEMM epilogue)
    gemm_kernel<<<dim3(DI_ / 64, (L_ * B_) / 64), 256, 0, stream>>>(oh, w1, f1h, b1, D_, DI_, 1);
    gemm_kernel<<<dim3(D_ / 64, (L_ * B_) / 64), 256, 0, stream>>>(f1h, w2, f2h, b2, DI_, D_, 0);
    ln_kernel<<<L_ * B_, 256, 0, stream>>>(f2h, oh, l2w, l2b, h);

    gemm_kernel<<<dim3(DI_ / 64, (M_ * B_) / 64), 256, 0, stream>>>(og, w1, f1g, b1, D_, DI_, 1);
    gemm_kernel<<<dim3(D_ / 64, (M_ * B_) / 64), 256, 0, stream>>>(f1g, w2, f2g, b2, DI_, D_, 0);
    ln_kernel<<<M_ * B_, 256, 0, stream>>>(f2g, og, l2w, l2b, g);
  }

  outT_kernel<<<M_ * B_, 256, 0, stream>>>(g, out);
}
